// RNN_82214263980689
// MI455X (gfx1250) — matrix operand order repaired and checked by reading, not compiled, not run on hardware
//
#include <hip/hip_runtime.h>
#include <hip/hip_bf16.h>
#include <math.h>
#include <stdint.h>

// Problem dims (fixed by reference)
#define Nn 32
#define Tt 256
#define Vv 32000
#define Ee 512
#define Hh 512

typedef __attribute__((ext_vector_type(2)))  float    v2f;
typedef __attribute__((ext_vector_type(8)))  float    v8f;
typedef __attribute__((ext_vector_type(16))) __bf16   v16bf;
typedef __attribute__((ext_vector_type(4)))  uint32_t v4u;
typedef __attribute__((ext_vector_type(8)))  int      v8i;
typedef __attribute__((ext_vector_type(4)))  int      v4i;

union BF16Frag { v16bf v; uint4 q[2]; };

__device__ __forceinline__ void lse_combine(float& m, float& s, float m2, float s2) {
  float M = fmaxf(m, m2);
  s = s * __expf(m - M) + s2 * __expf(m2 - M);
  m = M;
}

// ---------------------------------------------------------------- convert
__global__ void k_convert_bf16(const float* __restrict__ W, __bf16* __restrict__ Wb, int n) {
  int base = (blockIdx.x * blockDim.x + threadIdx.x) * 8;
  if (base + 7 < n) {
    float4 a = *(const float4*)(W + base);
    float4 b = *(const float4*)(W + base + 4);
    union { __bf16 h[8]; uint4 q; } o;
    o.h[0] = (__bf16)a.x; o.h[1] = (__bf16)a.y; o.h[2] = (__bf16)a.z; o.h[3] = (__bf16)a.w;
    o.h[4] = (__bf16)b.x; o.h[5] = (__bf16)b.y; o.h[6] = (__bf16)b.z; o.h[7] = (__bf16)b.w;
    *(uint4*)(Wb + base) = o.q;
  }
}

// ---------------------------------------------------------------- init
__global__ void k_init(float* __restrict__ h0, float* __restrict__ alpha) {
  int i = blockIdx.x * blockDim.x + threadIdx.x;
  for (int j = i; j < Nn * Hh; j += gridDim.x * blockDim.x) h0[j] = 0.f;
  if (i < Nn) alpha[i] = 0.f;
}

// ---------------------------------------------------------------- recurrent step (f32 WMMA)
// h_t = tanh(emb[x[:,t]] @ W_ih^T + h_{t-1} @ W_hh^T + b_h)
__global__ void k_rnn_step(const int* __restrict__ x, const float* __restrict__ embed,
                           const float* __restrict__ Wih, const float* __restrict__ Whh,
                           const float* __restrict__ bh, const float* __restrict__ hprev,
                           float* __restrict__ hcur, __bf16* __restrict__ hbf, int t) {
  const int lane  = threadIdx.x & 31;
  const int wave  = threadIdx.x >> 5;
  const int tile  = blockIdx.x * 8 + wave;     // 0..63
  const int mtile = tile >> 5;                 // 0..1 (batch)
  const int ntile = tile & 31;                 // 0..31 (hidden)
  const int g  = lane >> 4;
  const int li = lane & 15;
  const int mg = mtile * 16 + li;              // batch row for A fragment
  const int ng = ntile * 16 + li;              // hidden col for B fragment / D col

  const int tok = x[mg * Tt + t];
  const float* arow = embed + (size_t)tok * Ee;
  const float* brow = Wih   + (size_t)ng  * Ee;

  v8f acc = {};
  #pragma unroll 8
  for (int kb = 0; kb < Ee / 4; ++kb) {
    v2f a = *(const v2f*)(arow + kb * 4 + 2 * g);   // A: K = 2g, 2g+1
    v2f b = *(const v2f*)(brow + kb * 4 + 2 * g);   // B: K = 2g, 2g+1
    acc = __builtin_amdgcn_wmma_f32_16x16x4_f32(false, a, false, b, (short)0, acc, false, false);
  }
  const float* arow2 = hprev + (size_t)mg * Hh;
  const float* brow2 = Whh   + (size_t)ng * Hh;
  #pragma unroll 8
  for (int kb = 0; kb < Hh / 4; ++kb) {
    v2f a = *(const v2f*)(arow2 + kb * 4 + 2 * g);
    v2f b = *(const v2f*)(brow2 + kb * 4 + 2 * g);
    acc = __builtin_amdgcn_wmma_f32_16x16x4_f32(false, a, false, b, (short)0, acc, false, false);
  }
  const float bias = bh[ng];
  #pragma unroll
  for (int r = 0; r < 8; ++r) {                 // D: row M = mtile*16 + 8g + r, col ng
    float v = tanhf(acc[r] + bias);
    int row = mtile * 16 + g * 8 + r;
    hcur[row * Hh + ng] = v;
    hbf [row * Hh + ng] = (__bf16)v;
  }
}

// ---------------------------------------------------------------- emit GEMM + partial logsumexp (bf16 WMMA)
// One block = full M (both 16-row tiles), so each W_emit B-fragment feeds 2 WMMAs.
#define EMIT_TPW   5                       // 16-col tiles per wave per M-tile
#define EMIT_WAVES 8
#define EMIT_BLKX  50                      // 50 * 8 * 5 * 16 = 32000 = V
#define NPART (EMIT_BLKX * EMIT_WAVES)     // 400 wave-partials per row

__global__ void k_emit_logsumexp(const __bf16* __restrict__ hbf, const __bf16* __restrict__ Web,
                                 const float* __restrict__ bemit, float2* __restrict__ partials) {
  __shared__ __bf16 lds_a[32 * Hh];        // 32 KB: full h_t slab (32 x 512 bf16)

  // ---- Stage A slab into LDS via the Tensor Data Mover (TDM) ----
#if __has_builtin(__builtin_amdgcn_tensor_load_to_lds)
  if (threadIdx.x < 32) {                  // wave 0 issues the DMA + waits on TENSORcnt
    uint64_t ga  = (uint64_t)(uintptr_t)hbf;
    uint32_t lo  = (uint32_t)(uintptr_t)lds_a;   // flat LDS addr: low 32 bits = LDS byte offset
    v4u g0;
    g0.x = 1u;                                         // count=1, user descriptor
    g0.y = lo;                                         // lds_addr
    g0.z = (uint32_t)ga;                               // global_addr[31:0]
    g0.w = (uint32_t)((ga >> 32) & 0x1FFFFFFu) | (2u << 30);  // global_addr[56:32], type=2
    v8i g1;
    g1[0] = 0x00010000;        // workgroup_mask=0, data_size=1 (2 bytes)
    g1[1] = (int)(512u << 16); // tensor_dim0[15:0]=512 in bits[63:48]
    g1[2] = (int)(32u  << 16); // tensor_dim0[31:16]=0, tensor_dim1[15:0]=32
    g1[3] = (int)(512u << 16); // tensor_dim1[31:16]=0, tile_dim0=512
    g1[4] = 32;                // tile_dim1=32, tile_dim2=0
    g1[5] = 512;               // tensor_dim0_stride[31:0]=512
    g1[6] = 0;                 // stride msbs / tensor_dim1_stride lo
    g1[7] = 0;
    v4i gz4 = {0, 0, 0, 0};    // groups 2/3 unused (2D tensor)
    v8i gz8 = {0, 0, 0, 0, 0, 0, 0, 0};
    __builtin_amdgcn_tensor_load_to_lds(g0, g1, gz4, gz4, gz8, 0);
    __builtin_amdgcn_s_wait_tensorcnt(0);
  }
  __syncthreads();
#else
  {
    const uint4* src = (const uint4*)hbf;
    uint4* dst = (uint4*)lds_a;
    for (int i = threadIdx.x; i < (32 * Hh * 2) / 16; i += blockDim.x) dst[i] = src[i];
  }
  __syncthreads();
#endif

  const int lane = threadIdx.x & 31;
  const int wave = threadIdx.x >> 5;
  const int g  = lane >> 4;
  const int li = lane & 15;
  const int tile0 = blockIdx.x * (EMIT_WAVES * EMIT_TPW) + wave * EMIT_TPW;

  v8f acc0[EMIT_TPW] = {};
  v8f acc1[EMIT_TPW] = {};

  #pragma unroll 2
  for (int kb = 0; kb < Hh / 32; ++kb) {
    const int ka0 = kb * 32 + 8 * g;                   // A: runs K = kb*32+8g and +16
    BF16Frag a0, a1;
    a0.q[0] = *(const uint4*)(lds_a + li * Hh + ka0);
    a0.q[1] = *(const uint4*)(lds_a + li * Hh + ka0 + 16);
    a1.q[0] = *(const uint4*)(lds_a + (16 + li) * Hh + ka0);
    a1.q[1] = *(const uint4*)(lds_a + (16 + li) * Hh + ka0 + 16);
    const int kb0 = kb * 32 + 8 * g;                   // B: runs K = kb*32+8g and +16
    #pragma unroll
    for (int ti = 0; ti < EMIT_TPW; ++ti) {
      const int col = (tile0 + ti) * 16 + li;
      const __bf16* wr = Web + (size_t)col * Hh + kb0;
      BF16Frag b;
      b.q[0] = *(const uint4*)(wr);
      b.q[1] = *(const uint4*)(wr + 16);
      acc0[ti] = __builtin_amdgcn_wmma_f32_16x16x32_bf16(false, a0.v, false, b.v,
                                                         (short)0, acc0[ti], false, false);
      acc1[ti] = __builtin_amdgcn_wmma_f32_16x16x32_bf16(false, a1.v, false, b.v,
                                                         (short)0, acc1[ti], false, false);
    }
  }

  // ---- two-pass logsumexp epilogue over register-resident logits ----
  // pass 0: bias add in place
  #pragma unroll
  for (int ti = 0; ti < EMIT_TPW; ++ti) {
    const float be = bemit[(tile0 + ti) * 16 + li];
    #pragma unroll
    for (int r = 0; r < 8; ++r) { acc0[ti][r] += be; acc1[ti][r] += be; }
  }
  // pass 1: per-lane max over this wave's tiles
  float mr0[8], mr1[8];
  #pragma unroll
  for (int r = 0; r < 8; ++r) { mr0[r] = acc0[0][r]; mr1[r] = acc1[0][r]; }
  #pragma unroll
  for (int ti = 1; ti < EMIT_TPW; ++ti) {
    #pragma unroll
    for (int r = 0; r < 8; ++r) {
      mr0[r] = fmaxf(mr0[r], acc0[ti][r]);
      mr1[r] = fmaxf(mr1[r], acc1[ti][r]);
    }
  }
  // pass 2: single-exp sum against the local max
  float sr0[8], sr1[8];
  #pragma unroll
  for (int r = 0; r < 8; ++r) { sr0[r] = 0.f; sr1[r] = 0.f; }
  #pragma unroll
  for (int ti = 0; ti < EMIT_TPW; ++ti) {
    #pragma unroll
    for (int r = 0; r < 8; ++r) {
      sr0[r] += __expf(acc0[ti][r] - mr0[r]);
      sr1[r] += __expf(acc1[ti][r] - mr1[r]);
    }
  }
  // butterfly across the 16 column-lanes of each half-wave
  for (int mask = 1; mask < 16; mask <<= 1) {
    #pragma unroll
    for (int r = 0; r < 8; ++r) {
      float m2, s2;
      m2 = __shfl_xor(mr0[r], mask, 32); s2 = __shfl_xor(sr0[r], mask, 32);
      lse_combine(mr0[r], sr0[r], m2, s2);
      m2 = __shfl_xor(mr1[r], mask, 32); s2 = __shfl_xor(sr1[r], mask, 32);
      lse_combine(mr1[r], sr1[r], m2, s2);
    }
  }
  if (li == 0) {                          // lanes 0 (rows 0..7) and 16 (rows 8..15) of each M-tile
    const int wid = blockIdx.x * EMIT_WAVES + wave;
    #pragma unroll
    for (int r = 0; r < 8; ++r) {
      partials[(g * 8 + r) * NPART + wid]        = make_float2(mr0[r], sr0[r]);
      partials[(16 + g * 8 + r) * NPART + wid]   = make_float2(mr1[r], sr1[r]);
    }
  }
}

// ---------------------------------------------------------------- per-step reduce + target logp
__global__ void k_step_reduce(const float2* __restrict__ partials, const int* __restrict__ x,
                              const float* __restrict__ hF, const float* __restrict__ Wemit,
                              const float* __restrict__ bemit, float* __restrict__ alpha, int t) {
  const int lane = threadIdx.x & 31;
  const int n    = threadIdx.x >> 5;                    // blockDim = 1024 -> one wave per batch row
  float m = -INFINITY, s = 0.f;
  for (int p = lane; p < NPART; p += 32) {
    float2 P = partials[n * NPART + p];
    lse_combine(m, s, P.x, P.y);
  }
  for (int mask = 1; mask < 32; mask <<= 1) {
    float m2 = __shfl_xor(m, mask, 32);
    float s2 = __shfl_xor(s, mask, 32);
    lse_combine(m, s, m2, s2);
  }
  const float logZ = m + __logf(s);
  const int w = x[n * Tt + t + 1];                      // target token
  const float* hr = hF    + (size_t)n * Hh;
  const float* wr = Wemit + (size_t)w * Hh;
  float dot = 0.f;
  for (int k = lane; k < Hh; k += 32) dot += hr[k] * wr[k];
  for (int mask = 1; mask < 32; mask <<= 1) dot += __shfl_xor(dot, mask, 32);
  if (lane == 0) alpha[n] += dot + bemit[w] - logZ;
}

// ---------------------------------------------------------------- finalize
__global__ void k_finalize(const float* __restrict__ alpha, float* __restrict__ out) {
  float v = alpha[threadIdx.x];                         // launched with 32 threads
  for (int mask = 1; mask < 32; mask <<= 1) v += __shfl_xor(v, mask, 32);
  if (threadIdx.x == 0) out[0] = -v / (float)Nn;
}

// ---------------------------------------------------------------- launch
extern "C" void kernel_launch(void* const* d_in, const int* in_sizes, int n_in,
                              void* d_out, int out_size, void* d_ws, size_t ws_size,
                              hipStream_t stream) {
  const int*   x     = (const int*)  d_in[0];
  const float* embed = (const float*)d_in[1];
  const float* Wih   = (const float*)d_in[2];
  const float* Whh   = (const float*)d_in[3];
  const float* bh    = (const float*)d_in[4];
  const float* Wemit = (const float*)d_in[5];
  const float* bemit = (const float*)d_in[6];
  float* out = (float*)d_out;

  char* ws = (char*)d_ws;
  size_t off = 0;
  __bf16* Web = (__bf16*)(ws + off); off += (size_t)Vv * Hh * sizeof(__bf16);   // 32.77 MB
  float*  h0  = (float*) (ws + off); off += (size_t)Nn * Hh * sizeof(float);
  float*  h1  = (float*) (ws + off); off += (size_t)Nn * Hh * sizeof(float);
  __bf16* hbf = (__bf16*)(ws + off); off += (size_t)Nn * Hh * sizeof(__bf16);
  float2* partials = (float2*)(ws + off); off += (size_t)Nn * NPART * sizeof(float2);
  float*  alpha = (float*)(ws + off); off += 256;

  k_convert_bf16<<<(Vv * Hh) / (256 * 8), 256, 0, stream>>>(Wemit, Web, Vv * Hh);
  k_init<<<32, 256, 0, stream>>>(h0, alpha);

  for (int t = 0; t < Tt - 1; ++t) {
    const float* hp = (t & 1) ? h1 : h0;
    float*       hc = (t & 1) ? h0 : h1;
    k_rnn_step<<<8, 256, 0, stream>>>(x, embed, Wih, Whh, bh, hp, hc, hbf, t);
    k_emit_logsumexp<<<EMIT_BLKX, 256, 0, stream>>>(hbf, Web, bemit, partials);
    k_step_reduce<<<1, 1024, 0, stream>>>(partials, x, hc, Wemit, bemit, alpha, t);
  }
  k_finalize<<<1, 32, 0, stream>>>(alpha, out);
}
